// PT_20641612825139
// MI455X (gfx1250) — matrix operand order repaired and checked by reading, not compiled, not run on hardware
//
#include <hip/hip_runtime.h>

#define D_MODEL 1024
#define FF_DIM  2048
#define NLAYER  6
#define NOBJ    100
#define NPAIR   4950
#define BATCH   2
#define NTOK    (BATCH*NPAIR)   // 9900

typedef __attribute__((ext_vector_type(8)))  _Float16 v8h;
typedef __attribute__((ext_vector_type(16))) _Float16 v16h;
typedef __attribute__((ext_vector_type(8)))  float    v8f;

// ---------------------------------------------------------------------------
// GEMM: C[M,N] = act(A[M,K] @ W[K,N] + bias), f16 WMMA with f32 accumulation.
// Block = 256 threads (8 waves), tile 128x128, K-step 32, double-buffered LDS,
// next-tile global loads overlapped with WMMA compute.
// ---------------------------------------------------------------------------
#define BM 128
#define BN 128
#define BK 32

__global__ __launch_bounds__(256) void gemm_wmma_f16(
    const float* __restrict__ A, const float* __restrict__ W,
    const float* __restrict__ bias, float* __restrict__ C,
    int M, int K, int N, int relu)
{
  __shared__ __align__(64) _Float16 lA[2][BM * BK];   // [m][k]
  __shared__ __align__(64) _Float16 lB[2][BN * BK];   // [n][k]

  const int tid  = threadIdx.x;
  const int wave = tid >> 5;
  const int lane = tid & 31;
  const int half = lane >> 4;   // 0: lanes 0-15, 1: lanes 16-31
  const int l16  = lane & 15;
  const int bm = blockIdx.x * BM;
  const int bn = blockIdx.y * BN;

  v8f acc[8];
  #pragma unroll
  for (int i = 0; i < 8; i++)
    #pragma unroll
    for (int j = 0; j < 8; j++) acc[i][j] = 0.0f;

  // cooperative staging assignments
  const int ar = tid >> 1;              // A row within tile (0..127)
  const int ac = (tid & 1) << 4;        // A col base (0 or 16)
  const int kr = tid >> 3;              // W row within K-tile (0..31)
  const int nc = (tid & 7) << 4;        // W col base (0..112)
  // Clamp out-of-range rows to M-1: they produce garbage accumulators for
  // rows that are never stored (store is guarded), but staging stays branch-free.
  const int arow_g = (bm + ar < M) ? (bm + ar) : (M - 1);
  const float* aptr = A + (size_t)arow_g * K + ac;
  const float* wptr = W + (size_t)kr * N + bn + nc;

  float4 av[4], wv[4];
  #pragma unroll
  for (int j = 0; j < 4; j++) av[j] = ((const float4*)aptr)[j];
  #pragma unroll
  for (int j = 0; j < 4; j++) wv[j] = ((const float4*)wptr)[j];

  // store tile 0 into buffer 0
  {
    _Float16* la = &lA[0][ar * BK + ac];
    #pragma unroll
    for (int j = 0; j < 4; j++) {
      la[j*4+0] = (_Float16)av[j].x; la[j*4+1] = (_Float16)av[j].y;
      la[j*4+2] = (_Float16)av[j].z; la[j*4+3] = (_Float16)av[j].w;
    }
    #pragma unroll
    for (int j = 0; j < 4; j++) {
      lB[0][(nc + j*4 + 0) * BK + kr] = (_Float16)wv[j].x;
      lB[0][(nc + j*4 + 1) * BK + kr] = (_Float16)wv[j].y;
      lB[0][(nc + j*4 + 2) * BK + kr] = (_Float16)wv[j].z;
      lB[0][(nc + j*4 + 3) * BK + kr] = (_Float16)wv[j].w;
    }
  }

  int buf = 0;
  const int arow_l = wave * 16 + l16;
  for (int k0 = 0; k0 < K; k0 += BK) {
    __syncthreads();
    const bool more = (k0 + BK) < K;
    // issue next tile's global loads early; they complete under the WMMAs
    if (more) {
      #pragma unroll
      for (int j = 0; j < 4; j++) av[j] = ((const float4*)(aptr + (k0 + BK)))[j];
      #pragma unroll
      for (int j = 0; j < 4; j++) wv[j] = ((const float4*)(wptr + (size_t)(k0 + BK) * N))[j];
    }

    // ---- A fragment: 16x32 f16, contiguous k = 16*half + s (same as B) ----
    const _Float16* La = lA[buf];
    const _Float16* Lb = lB[buf];
    v8h alo = *(const v8h*)&La[arow_l * BK + half * 16];
    v8h ahi = *(const v8h*)&La[arow_l * BK + half * 16 + 8];
    v16h afrag = __builtin_shufflevector(alo, ahi,
        0,1,2,3,4,5,6,7,8,9,10,11,12,13,14,15);

    // preload all 8 B fragments, then run the WMMA chain
    v16h bfrag[8];
    #pragma unroll
    for (int nt = 0; nt < 8; nt++)
      bfrag[nt] = *(const v16h*)&Lb[(nt * 16 + l16) * BK + half * 16];
    #pragma unroll
    for (int nt = 0; nt < 8; nt++)
      acc[nt] = __builtin_amdgcn_wmma_f32_16x16x32_f16(
          false, afrag, false, bfrag[nt], (short)0, acc[nt], false, false);

    // stage next tile into the other buffer (current buffer still being read
    // is untouched; barrier at loop top orders visibility)
    if (more) {
      const int nb = buf ^ 1;
      _Float16* la = &lA[nb][ar * BK + ac];
      #pragma unroll
      for (int j = 0; j < 4; j++) {
        la[j*4+0] = (_Float16)av[j].x; la[j*4+1] = (_Float16)av[j].y;
        la[j*4+2] = (_Float16)av[j].z; la[j*4+3] = (_Float16)av[j].w;
      }
      #pragma unroll
      for (int j = 0; j < 4; j++) {
        lB[nb][(nc + j*4 + 0) * BK + kr] = (_Float16)wv[j].x;
        lB[nb][(nc + j*4 + 1) * BK + kr] = (_Float16)wv[j].y;
        lB[nb][(nc + j*4 + 2) * BK + kr] = (_Float16)wv[j].z;
        lB[nb][(nc + j*4 + 3) * BK + kr] = (_Float16)wv[j].w;
      }
      buf = nb;
    }
  }

  // ---- epilogue: C/D layout: VGPR r -> M = half*8 + r, N = l16 ----
  const int rbase = bm + wave * 16 + half * 8;
  #pragma unroll
  for (int nt = 0; nt < 8; nt++) {
    const int col = bn + nt * 16 + l16;
    const float bb = bias[col];
    #pragma unroll
    for (int r = 0; r < 8; r++) {
      int row = rbase + r;
      if (row < M) {
        float vo = acc[nt][r] + bb;
        if (relu) vo = fmaxf(vo, 0.0f);
        C[(size_t)row * N + col] = vo;
      }
    }
  }
}

// ---------------------------------------------------------------------------
// Fused residual-add + LayerNorm over D=1024. One wave32 per row, b128 I/O.
// ---------------------------------------------------------------------------
__global__ __launch_bounds__(256) void add_ln(
    const float* __restrict__ X, const float* __restrict__ R,
    const float* __restrict__ w, const float* __restrict__ b,
    float* __restrict__ O, int M, int has_res)
{
  const int row  = blockIdx.x * 8 + (threadIdx.x >> 5);
  const int lane = threadIdx.x & 31;
  if (row >= M) return;
  const float* x = X + (size_t)row * D_MODEL;
  const float* r = R + (size_t)row * D_MODEL;
  float4 v[8];
  float s = 0.f;
  #pragma unroll
  for (int i = 0; i < 8; i++) {
    const int c = i * 128 + lane * 4;
    float4 t = *(const float4*)&x[c];
    if (has_res) {
      float4 rr = *(const float4*)&r[c];
      t.x += rr.x; t.y += rr.y; t.z += rr.z; t.w += rr.w;
    }
    v[i] = t; s += t.x + t.y + t.z + t.w;
  }
  #pragma unroll
  for (int off = 16; off > 0; off >>= 1) s += __shfl_xor(s, off, 32);
  const float mu = s * (1.0f / D_MODEL);
  float s2 = 0.f;
  #pragma unroll
  for (int i = 0; i < 8; i++) {
    float dx = v[i].x - mu, dy = v[i].y - mu, dz = v[i].z - mu, dw = v[i].w - mu;
    s2 += dx*dx + dy*dy + dz*dz + dw*dw;
  }
  #pragma unroll
  for (int off = 16; off > 0; off >>= 1) s2 += __shfl_xor(s2, off, 32);
  const float rs = rsqrtf(s2 * (1.0f / D_MODEL) + 1e-5f);
  #pragma unroll
  for (int i = 0; i < 8; i++) {
    const int c = i * 128 + lane * 4;
    float4 ww = *(const float4*)&w[c];
    float4 bb = *(const float4*)&b[c];
    float4 o;
    o.x = (v[i].x - mu) * rs * ww.x + bb.x;
    o.y = (v[i].y - mu) * rs * ww.y + bb.y;
    o.z = (v[i].z - mu) * rs * ww.z + bb.z;
    o.w = (v[i].w - mu) * rs * ww.w + bb.w;
    *(float4*)&O[(size_t)row * D_MODEL + c] = o;
  }
}

// ---------------------------------------------------------------------------
// Prologue: pairwise pos features (N x 5) and feat = (x_j + x_k)/2 (N x D).
// ---------------------------------------------------------------------------
__global__ __launch_bounds__(256) void pairwise_feat(
    const float* __restrict__ x, const float* __restrict__ z,
    float* __restrict__ pos, float* __restrict__ feat)
{
  const int n   = blockIdx.x;      // token 0..NTOK-1
  const int tid = threadIdx.x;
  const int bb  = n / NPAIR;
  const int p   = n % NPAIR;
  __shared__ int sj, sk;
  if (tid == 0) {
    int j = 0, rem = p;
    while (rem >= (NOBJ - 1 - j)) { rem -= (NOBJ - 1 - j); j++; }
    int k = j + 1 + rem;
    sj = j; sk = k;
    const float* zj = z + ((size_t)bb * NOBJ + j) * 5;
    const float* zk = z + ((size_t)bb * NOBJ + k) * 5;
    float xj = 0.5f * (zj[0] + zj[2]), xk = 0.5f * (zk[0] + zk[2]);
    float yj = 0.5f * (zj[1] + zj[3]), yk = 0.5f * (zk[1] + zk[3]);
    float dx = xj - xk, dy = yj - yk;
    float* pp = pos + (size_t)n * 5;
    pp[0] = (dx > 0.f) ? 1.f : ((dx < 0.f) ? -1.f : 0.f);
    pp[1] = (dy > 0.f) ? 1.f : ((dy < 0.f) ? -1.f : 0.f);
    pp[2] = sqrtf(dx * dx + dy * dy);
    pp[3] = zj[4];
    pp[4] = zk[4];
  }
  __syncthreads();
  const float* xj = x + ((size_t)bb * NOBJ + sj) * D_MODEL;
  const float* xk = x + ((size_t)bb * NOBJ + sk) * D_MODEL;
  const int d = tid * 4;           // 256 threads * 4 = 1024 exactly
  float4 a = *(const float4*)&xj[d];
  float4 c = *(const float4*)&xk[d];
  float4 o;
  o.x = 0.5f * (a.x + c.x); o.y = 0.5f * (a.y + c.y);
  o.z = 0.5f * (a.z + c.z); o.w = 0.5f * (a.w + c.w);
  *(float4*)&feat[(size_t)n * D_MODEL + d] = o;
}

// hid = relu(pos @ w1 + b1), K=5 -> plain VALU kernel
__global__ __launch_bounds__(256) void mlp_hidden(
    const float* __restrict__ pos, const float* __restrict__ w1,
    const float* __restrict__ b1, float* __restrict__ hid)
{
  int idx = blockIdx.x * 256 + threadIdx.x;
  if (idx >= NTOK * 512) return;
  int n = idx >> 9, u = idx & 511;
  const float* pp = pos + (size_t)n * 5;
  float s = b1[u];
  #pragma unroll
  for (int i = 0; i < 5; i++) s += pp[i] * w1[i * 512 + u];
  hid[idx] = fmaxf(s, 0.f);
}

// ---------------------------------------------------------------------------
extern "C" void kernel_launch(void* const* d_in, const int* in_sizes, int n_in,
                              void* d_out, int out_size, void* d_ws, size_t ws_size,
                              hipStream_t stream)
{
  (void)in_sizes; (void)n_in; (void)out_size; (void)ws_size;
  const float* x         = (const float*)d_in[0];
  const float* z         = (const float*)d_in[1];
  const float* mlp_w1    = (const float*)d_in[2];
  const float* mlp_b1    = (const float*)d_in[3];
  const float* mlp_w2    = (const float*)d_in[4];
  const float* mlp_b2    = (const float*)d_in[5];
  const float* enc_qkv_w = (const float*)d_in[6];
  const float* enc_qkv_b = (const float*)d_in[7];
  const float* enc_out_w = (const float*)d_in[8];
  const float* enc_out_b = (const float*)d_in[9];
  const float* enc_ff1_w = (const float*)d_in[10];
  const float* enc_ff1_b = (const float*)d_in[11];
  const float* enc_ff2_w = (const float*)d_in[12];
  const float* enc_ff2_b = (const float*)d_in[13];
  const float* enc_ln_w  = (const float*)d_in[14];
  const float* enc_ln_b  = (const float*)d_in[15];
  const float* enc_fln_w = (const float*)d_in[16];
  const float* enc_fln_b = (const float*)d_in[17];
  const float* dec_sqkv_w= (const float*)d_in[18];
  const float* dec_sqkv_b= (const float*)d_in[19];
  const float* dec_sout_w= (const float*)d_in[20];
  const float* dec_sout_b= (const float*)d_in[21];
  const float* dec_cqkv_w= (const float*)d_in[22];
  const float* dec_cqkv_b= (const float*)d_in[23];
  const float* dec_cout_w= (const float*)d_in[24];
  const float* dec_cout_b= (const float*)d_in[25];
  const float* dec_ff1_w = (const float*)d_in[26];
  const float* dec_ff1_b = (const float*)d_in[27];
  const float* dec_ff2_w = (const float*)d_in[28];
  const float* dec_ff2_b = (const float*)d_in[29];
  const float* dec_ln_w  = (const float*)d_in[30];
  const float* dec_ln_b  = (const float*)d_in[31];
  const float* dec_fln_w = (const float*)d_in[32];
  const float* dec_fln_b = (const float*)d_in[33];

  float* ws = (float*)d_ws;
  size_t off = 0;
  auto alloc = [&](size_t nelem) {
    float* p = ws + off; off += (nelem + 63) & ~(size_t)63; return p;
  };
  float* posb = alloc((size_t)NTOK * 5);
  float* hid  = alloc((size_t)NTOK * 512);
  float* h    = alloc((size_t)NTOK * D_MODEL);
  float* t    = alloc((size_t)NTOK * D_MODEL);
  float* mem  = alloc((size_t)NTOK * D_MODEL);
  float* tmp2 = alloc((size_t)NTOK * D_MODEL);
  float* ffb  = alloc((size_t)NTOK * FF_DIM);
  float* tmp1 = ffb;   // aliased: attention scratch and FFN buffer never live together

  auto gemm = [&](const float* A, const float* Wt, const float* bias, float* Cc,
                  int M, int K, int N, int relu) {
    dim3 g((unsigned)((M + BM - 1) / BM), (unsigned)(N / BN));
    gemm_wmma_f16<<<g, 256, 0, stream>>>(A, Wt, bias, Cc, M, K, N, relu);
  };
  auto lnadd = [&](const float* X, const float* R, const float* w, const float* b,
                   float* O, int has_res) {
    add_ln<<<(NTOK + 7) / 8, 256, 0, stream>>>(X, R, w, b, O, NTOK, has_res);
  };

  // prologue: pos + feat (feat initializes decoder stream t)
  pairwise_feat<<<NTOK, 256, 0, stream>>>(x, z, posb, t);
  mlp_hidden<<<(NTOK * 512 + 255) / 256, 256, 0, stream>>>(posb, mlp_w1, mlp_b1, hid);
  gemm(hid, mlp_w2, mlp_b2, h, NTOK, 512, D_MODEL, 0);

  // encoder: seq-len-1 attention collapses to V then O projection
  for (int i = 0; i < NLAYER; i++) {
    gemm(h, enc_qkv_w + (size_t)(i*3+2)*D_MODEL*D_MODEL,
            enc_qkv_b + (size_t)(i*3+2)*D_MODEL, tmp1, NTOK, D_MODEL, D_MODEL, 0);
    gemm(tmp1, enc_out_w + (size_t)i*D_MODEL*D_MODEL,
               enc_out_b + (size_t)i*D_MODEL, tmp2, NTOK, D_MODEL, D_MODEL, 0);
    lnadd(h, tmp2, enc_ln_w + (size_t)(i*2)*D_MODEL, enc_ln_b + (size_t)(i*2)*D_MODEL, h, 1);
    gemm(h, enc_ff1_w + (size_t)i*D_MODEL*FF_DIM,
            enc_ff1_b + (size_t)i*FF_DIM, ffb, NTOK, D_MODEL, FF_DIM, 1);
    gemm(ffb, enc_ff2_w + (size_t)i*FF_DIM*D_MODEL,
              enc_ff2_b + (size_t)i*D_MODEL, tmp2, NTOK, FF_DIM, D_MODEL, 0);
    lnadd(h, tmp2, enc_ln_w + (size_t)(i*2+1)*D_MODEL, enc_ln_b + (size_t)(i*2+1)*D_MODEL, h, 1);
  }
  lnadd(h, h, enc_fln_w, enc_fln_b, mem, 0);

  // decoder
  for (int i = 0; i < NLAYER; i++) {
    gemm(t, dec_sqkv_w + (size_t)(i*3+2)*D_MODEL*D_MODEL,
            dec_sqkv_b + (size_t)(i*3+2)*D_MODEL, tmp1, NTOK, D_MODEL, D_MODEL, 0);
    gemm(tmp1, dec_sout_w + (size_t)i*D_MODEL*D_MODEL,
               dec_sout_b + (size_t)i*D_MODEL, tmp2, NTOK, D_MODEL, D_MODEL, 0);
    lnadd(t, tmp2, dec_ln_w + (size_t)(i*3)*D_MODEL, dec_ln_b + (size_t)(i*3)*D_MODEL, t, 1);
    gemm(mem, dec_cqkv_w + (size_t)(i*3+2)*D_MODEL*D_MODEL,
              dec_cqkv_b + (size_t)(i*3+2)*D_MODEL, tmp1, NTOK, D_MODEL, D_MODEL, 0);
    gemm(tmp1, dec_cout_w + (size_t)i*D_MODEL*D_MODEL,
               dec_cout_b + (size_t)i*D_MODEL, tmp2, NTOK, D_MODEL, D_MODEL, 0);
    lnadd(t, tmp2, dec_ln_w + (size_t)(i*3+1)*D_MODEL, dec_ln_b + (size_t)(i*3+1)*D_MODEL, t, 1);
    gemm(t, dec_ff1_w + (size_t)i*D_MODEL*FF_DIM,
            dec_ff1_b + (size_t)i*FF_DIM, ffb, NTOK, D_MODEL, FF_DIM, 1);
    gemm(ffb, dec_ff2_w + (size_t)i*FF_DIM*D_MODEL,
              dec_ff2_b + (size_t)i*D_MODEL, tmp2, NTOK, FF_DIM, D_MODEL, 0);
    lnadd(t, tmp2, dec_ln_w + (size_t)(i*3+2)*D_MODEL, dec_ln_b + (size_t)(i*3+2)*D_MODEL, t, 1);
  }
  lnadd(t, t, dec_fln_w, dec_fln_b, (float*)d_out, 0);
}
